// DeformableTransformer_1159641170359
// MI455X (gfx1250) — matrix operand order repaired and checked by reading, not compiled, not run on hardware
//
#include <hip/hip_runtime.h>

#define B_   2
#define LQ_  12240
#define DM_  256
#define M_   8
#define L_   4
#define P_   4
#define D_   32
#define ROWS (B_ * LQ_)   // 24480

typedef unsigned short u16;
typedef __attribute__((ext_vector_type(8)))  u16    u16x8;
typedef __attribute__((ext_vector_type(16))) u16    u16x16;
typedef __attribute__((ext_vector_type(16))) __bf16 v16bf;
typedef __attribute__((ext_vector_type(8)))  float  v8f;

__device__ __forceinline__ u16 f2bf(float f) {
    unsigned u = __float_as_uint(f);
    u += 0x7FFFu + ((u >> 16) & 1u);   // round-to-nearest-even
    return (u16)(u >> 16);
}

// load a 16-element bf16 fragment: 8 elems at p, 8 at p+hioff (two 16B loads)
__device__ __forceinline__ v16bf ldfrag(const u16* __restrict__ p, int hioff) {
    u16x8 lo = *(const u16x8*)p;
    u16x8 hi = *(const u16x8*)(p + hioff);
    u16x16 v = __builtin_shufflevector(lo, hi, 0,1,2,3,4,5,6,7,8,9,10,11,12,13,14,15);
    return __builtin_bit_cast(v16bf, v);
}

// ---------------- elementwise f32 -> bf16 ----------------
__global__ __launch_bounds__(256) void cvt_bf16_kernel(const float* __restrict__ in,
                                                       u16* __restrict__ out, int n) {
    int i = blockIdx.x * 256 + threadIdx.x;
    if (i < n) out[i] = f2bf(in[i]);
}

// ---------------- weight transpose + convert: [K x N] f32 -> [N x K] bf16 ----------------
__global__ __launch_bounds__(256) void cvt_T_kernel(const float* __restrict__ in,
                                                    u16* __restrict__ out, int K, int N) {
    int i = blockIdx.x * 256 + threadIdx.x;
    if (i < K * N) {
        int n = i / K, k = i - n * K;
        out[i] = f2bf(in[k * N + n]);
    }
}

// ---------------- bf16 WMMA GEMM: C[ROWS x N] = A[ROWS x 256] * B[256 x N] + bias ----------------
// A row-major bf16, BT is B transposed (N x 256) bf16, C f32. One wave -> 32x64 tile.
template <int N>
__global__ __launch_bounds__(256) void gemm_bf16_kernel(const u16* __restrict__ A,
                                                        const u16* __restrict__ BT,
                                                        const float* __restrict__ bias,
                                                        float* __restrict__ C) {
    constexpr int K      = 256;
    constexpr int tilesN = N >> 6;
    constexpr int total  = (ROWS >> 5) * tilesN;
    int wid  = (blockIdx.x * 256 + threadIdx.x) >> 5;
    int lane = threadIdx.x & 31;
    if (wid >= total) return;                  // wave-uniform: EXEC stays all-ones for WMMA
    int jn = wid % tilesN, im = wid / tilesN;
    int i0 = im << 5, j0 = jn << 6;
    int half = lane >> 4, l15 = lane & 15;

    // A fragment: row = i0 + (lane&15); lanes>=16 hold K {8..15}u{24..31} of each 32-chunk
    const u16* aptr = A + (size_t)(i0 + l15) * K + half * 8;
    // B fragment: col = j0 + (lane&15); lanes>=16 hold K {8..15}u{24..31} of each 32-chunk
    const u16* bptr = BT + (size_t)(j0 + l15) * K + half * 8;

    v8f acc[2][4];
#pragma unroll
    for (int jj = 0; jj < 4; ++jj) {
        float bb = bias[j0 + jj * 16 + l15];
        v8f c = (v8f){bb, bb, bb, bb, bb, bb, bb, bb};
        acc[0][jj] = c;
        acc[1][jj] = c;
    }

#pragma unroll
    for (int k0 = 0; k0 < K; k0 += 32) {
        v16bf a0 = ldfrag(aptr + k0, 16);               // rows i0..i0+15
        v16bf a1 = ldfrag(aptr + (size_t)16 * K + k0, 16); // rows i0+16..i0+31
#pragma unroll
        for (int jj = 0; jj < 4; ++jj) {
            v16bf b = ldfrag(bptr + (size_t)jj * 16 * K + k0, 16);
            acc[0][jj] = __builtin_amdgcn_wmma_f32_16x16x32_bf16(false, a0, false, b,
                                                                 (short)0, acc[0][jj], false, false);
            acc[1][jj] = __builtin_amdgcn_wmma_f32_16x16x32_bf16(false, a1, false, b,
                                                                 (short)0, acc[1][jj], false, false);
        }
    }

#pragma unroll
    for (int sub = 0; sub < 2; ++sub) {
        int orow = i0 + sub * 16 + half * 8;
#pragma unroll
        for (int r = 0; r < 8; ++r) {
#pragma unroll
            for (int jj = 0; jj < 4; ++jj)
                C[(size_t)(orow + r) * N + j0 + jj * 16 + l15] = acc[sub][jj][r];
        }
    }
}

// ---------------- deformable sampling: one wave per (b, q, head), lane = channel ----------------
__global__ __launch_bounds__(256) void sample_kernel(const float* __restrict__ value,  // (B*LQ, 256)
                                                     const float* __restrict__ refpt,  // (B,LQ,L,2)
                                                     const float* __restrict__ offs,   // (B,LQ,256)
                                                     const float* __restrict__ attn,   // (B,LQ,128)
                                                     u16* __restrict__ sampled) {      // (B,LQ,256) bf16
    int wid  = (blockIdx.x * 256 + threadIdx.x) >> 5;
    int lane = threadIdx.x & 31;
    if (wid >= ROWS * M_) return;
    int m  = wid & (M_ - 1);
    int bq = wid >> 3;
    int b  = bq / LQ_;

    // per-wave operand preloads (coalesced across lanes)
    float offv  = offs[(size_t)bq * 256 + m * 32 + lane];                     // 32 offset scalars
    float logit = (lane < 16) ? attn[(size_t)bq * 128 + m * 16 + lane] : -3.0e38f;
    float rv    = (lane < 8) ? refpt[(size_t)bq * 8 + lane] : 0.0f;

    // softmax over 16 logits (xor masks < 16 stay inside each 16-lane half)
    float mx = logit;
    for (int s = 8; s; s >>= 1) mx = fmaxf(mx, __shfl_xor(mx, s, 32));
    float e = (lane < 16) ? __expf(logit - mx) : 0.0f;
    float sum = e;
    for (int s = 8; s; s >>= 1) sum += __shfl_xor(sum, s, 32);
    float w = e / sum;

    const int Hs[4]  = {96, 48, 24, 12};
    const int Ws_[4] = {96, 48, 24, 12};
    const int st_[4] = {0, 9216, 11520, 12096};

    const float* vbase = value + (size_t)b * LQ_ * 256 + m * 32 + lane;
    float acc = 0.0f;

#pragma unroll
    for (int l = 0; l < 4; ++l) {
        const int H = Hs[l], W = Ws_[l], st = st_[l];
        float rx = __shfl(rv, 2 * l, 32);
        float ry = __shfl(rv, 2 * l + 1, 32);
#pragma unroll
        for (int p = 0; p < 4; ++p) {
            float ox = __shfl(offv, (l * 4 + p) * 2, 32);
            float oy = __shfl(offv, (l * 4 + p) * 2 + 1, 32);
            float lx = rx + ox / (float)W;
            float ly = ry + oy / (float)H;
            float x = lx * (float)W - 0.5f;
            float y = ly * (float)H - 0.5f;
            float x0f = floorf(x), y0f = floorf(y);
            float fx = x - x0f, fy = y - y0f;
            int x0 = (int)x0f, y0 = (int)y0f;
            float wa = __shfl(w, l * 4 + p, 32);

            float w00 = (1.0f - fx) * (1.0f - fy) * wa;
            float w01 = fx * (1.0f - fy) * wa;
            float w10 = (1.0f - fx) * fy * wa;
            float w11 = fx * fy * wa;
            // zero weights of invalid corners (reference semantics)
            if (x0 < 0 || x0 >= W) { w00 = 0.0f; w10 = 0.0f; }
            if (x0 + 1 < 0 || x0 + 1 >= W) { w01 = 0.0f; w11 = 0.0f; }
            if (y0 < 0 || y0 >= H) { w00 = 0.0f; w01 = 0.0f; }
            if (y0 + 1 < 0 || y0 + 1 >= H) { w10 = 0.0f; w11 = 0.0f; }

            int cx0 = min(max(x0, 0), W - 1), cx1 = min(max(x0 + 1, 0), W - 1);
            int cy0 = min(max(y0, 0), H - 1), cy1 = min(max(y0 + 1, 0), H - 1);

            float s00 = vbase[(size_t)(st + cy0 * W + cx0) * 256];
            float s01 = vbase[(size_t)(st + cy0 * W + cx1) * 256];
            float s10 = vbase[(size_t)(st + cy1 * W + cx0) * 256];
            float s11 = vbase[(size_t)(st + cy1 * W + cx1) * 256];
            acc += w00 * s00 + w01 * s01 + w10 * s10 + w11 * s11;
        }
    }
    sampled[(size_t)bq * 256 + m * 32 + lane] = f2bf(acc);
}

extern "C" void kernel_launch(void* const* d_in, const int* in_sizes, int n_in,
                              void* d_out, int out_size, void* d_ws, size_t ws_size,
                              hipStream_t stream) {
    const float* query = (const float*)d_in[0];
    const float* refpt = (const float*)d_in[1];
    const float* input = (const float*)d_in[2];
    // d_in[3] spatial_shapes: compile-time constants used instead
    const float* Wv = (const float*)d_in[4];
    const float* bv = (const float*)d_in[5];
    const float* Ws = (const float*)d_in[6];
    const float* bs = (const float*)d_in[7];
    const float* Wa = (const float*)d_in[8];
    const float* ba = (const float*)d_in[9];
    const float* Wo = (const float*)d_in[10];
    const float* bo = (const float*)d_in[11];
    float* out = (float*)d_out;

    char* ws = (char*)d_ws;
    size_t off = 0;
    auto alloc = [&](size_t bytes) -> void* {
        void* p = ws + off;
        off = (off + bytes + 255) & ~(size_t)255;
        return p;
    };

    u16*   qb   = (u16*)alloc((size_t)ROWS * 256 * 2);
    u16*   ib   = (u16*)alloc((size_t)ROWS * 256 * 2);
    u16*   WvT  = (u16*)alloc(256 * 256 * 2);
    u16*   WsT  = (u16*)alloc(256 * 256 * 2);
    u16*   WaT  = (u16*)alloc(128 * 256 * 2);
    u16*   WoT  = (u16*)alloc(256 * 256 * 2);
    float* val  = (float*)alloc((size_t)ROWS * 256 * 4);
    float* offr = (float*)alloc((size_t)ROWS * 256 * 4);
    float* attr = (float*)alloc((size_t)ROWS * 128 * 4);
    u16*   samp = (u16*)alloc((size_t)ROWS * 256 * 2);

    const int nElem = ROWS * 256;               // 6,266,880
    cvt_bf16_kernel<<<(nElem + 255) / 256, 256, 0, stream>>>(query, qb, nElem);
    cvt_bf16_kernel<<<(nElem + 255) / 256, 256, 0, stream>>>(input, ib, nElem);
    cvt_T_kernel<<<(256 * 256 + 255) / 256, 256, 0, stream>>>(Wv, WvT, 256, 256);
    cvt_T_kernel<<<(256 * 256 + 255) / 256, 256, 0, stream>>>(Ws, WsT, 256, 256);
    cvt_T_kernel<<<(256 * 128 + 255) / 256, 256, 0, stream>>>(Wa, WaT, 256, 128);
    cvt_T_kernel<<<(256 * 256 + 255) / 256, 256, 0, stream>>>(Wo, WoT, 256, 256);

    // total waves = (ROWS/32) * (N/64); 8 waves per block
    const int blk256 = ((ROWS / 32) * (256 / 64) + 7) / 8;   // 383
    const int blk128 = ((ROWS / 32) * (128 / 64) + 7) / 8;   // 192
    gemm_bf16_kernel<256><<<blk256, 256, 0, stream>>>(ib, WvT, bv, val);
    gemm_bf16_kernel<256><<<blk256, 256, 0, stream>>>(qb, WsT, bs, offr);
    gemm_bf16_kernel<128><<<blk128, 256, 0, stream>>>(qb, WaT, ba, attr);

    sample_kernel<<<(ROWS * M_) / 8, 256, 0, stream>>>(val, refpt, offr, attr, samp);

    gemm_bf16_kernel<256><<<blk256, 256, 0, stream>>>(samp, WoT, bo, out);
}
